// GCU_77077483094500
// MI455X (gfx1250) — matrix operand order repaired and checked by reading, not compiled, not run on hardware
//
#include <hip/hip_runtime.h>

// ---------------------------------------------------------------------------
// Fused GNN block for MI455X (gfx1250, wave32, WMMA f32_16x16x32_f16)
//   kernel 1: neighbor MLP  (~63 GFLOP)  -> v_ = sum_k tanh-MLP([z|z_k])
//   kernel 2: current MLP + output head  -> dc = [v, v_] @ out_w^T + out_b
// nt-outer loops keep ONE 8-VGPR accumulator live; B-tiles stream from LDS
// (ds_load_b128 -> v_wmma); tanh uses the CDNA5 V_TANH_F32 trans op.
// ---------------------------------------------------------------------------

typedef __attribute__((ext_vector_type(16))) _Float16 v16h;
typedef __attribute__((ext_vector_type(8)))  _Float16 v8h;
typedef __attribute__((ext_vector_type(8)))  float    v8f;

#define DIM_D   128
#define DIM_H   128
#define DIM_C   64
#define KNBR    32
#define HS      136   // padded f16 row stride for staged activations

__device__ __forceinline__ v8f vzero8() {
  v8f z = {0.f, 0.f, 0.f, 0.f, 0.f, 0.f, 0.f, 0.f};
  return z;
}

// CDNA5 hardware tanh (TRANS32). ISA 7.4: one independent op / V_NOP must
// follow a TRANS op before its output is consumed -> include it in the asm.
__device__ __forceinline__ float fast_tanh(float x) {
  float r;
  asm volatile("v_tanh_f32 %0, %1\n\tv_nop" : "=v"(r) : "v"(x));
  return r;
}

// D = A(16x32) * B(32x16) + C, f16 inputs, f32 accumulate
__device__ __forceinline__ v8f wmma_f16(v16h a, v16h b, v8f c) {
  return __builtin_amdgcn_wmma_f32_16x16x32_f16(
      false, a, false, b, (short)0, c, false, false);
}

// ---- A-fragment loaders --------------------------------------------------
// ISA layout (16-bit A 16x32): lane<16 row M=lane holds K = kbase+{0..7,16..23}
//                              lane>=16 row M=lane-16 holds K = kbase+{8..15,24..31}
__device__ __forceinline__ v16h load_a_f32(const float* __restrict__ row,
                                           int kbase, int lane) {
  const int k0 = kbase + ((lane & 16) ? 8 : 0);
  const float4 x0 = *(const float4*)(row + k0);
  const float4 x1 = *(const float4*)(row + k0 + 4);
  const float4 y0 = *(const float4*)(row + k0 + 16);
  const float4 y1 = *(const float4*)(row + k0 + 20);
  v16h a;
  a[0]  = (_Float16)x0.x; a[1]  = (_Float16)x0.y;
  a[2]  = (_Float16)x0.z; a[3]  = (_Float16)x0.w;
  a[4]  = (_Float16)x1.x; a[5]  = (_Float16)x1.y;
  a[6]  = (_Float16)x1.z; a[7]  = (_Float16)x1.w;
  a[8]  = (_Float16)y0.x; a[9]  = (_Float16)y0.y;
  a[10] = (_Float16)y0.z; a[11] = (_Float16)y0.w;
  a[12] = (_Float16)y1.x; a[13] = (_Float16)y1.y;
  a[14] = (_Float16)y1.z; a[15] = (_Float16)y1.w;
  return a;
}

__device__ __forceinline__ v16h load_a_lds(const _Float16* row,
                                           int kbase, int lane) {
  const int k0 = kbase + ((lane & 16) ? 8 : 0);
  const v8h lo = *(const v8h*)(row + k0);       // ds_load_b128
  const v8h hi = *(const v8h*)(row + k0 + 16);  // ds_load_b128
  v16h a;
#pragma unroll
  for (int i = 0; i < 8; ++i) { a[i] = lo[i]; a[8 + i] = hi[i]; }
  return a;
}

// ---- B-fragment loader ---------------------------------------------------
// B(32x16) with B[k][n] = W[n][k] (W row-major [out][in], f16 in LDS):
// lane holds column N=lane&15, K = kbase + (lane>=16?8:0) + {0..7,16..23}
__device__ __forceinline__ v16h load_b_lds(const _Float16* W, int ldw,
                                           int ncol, int kbase, int lane) {
  const int n  = ncol + (lane & 15);
  const int k0 = kbase + ((lane & 16) ? 8 : 0);
  const _Float16* p = W + n * ldw + k0;
  const v8h lo = *(const v8h*)(p);
  const v8h hi = *(const v8h*)(p + 16);
  v16h b;
#pragma unroll
  for (int i = 0; i < 8; ++i) { b[i] = lo[i]; b[8 + i] = hi[i]; }
  return b;
}

// ===========================================================================
// Kernel 1: neighbor branch.  One wave per node, 8 waves (256 thr) per WG.
// ===========================================================================
__global__ __launch_bounds__(256)
__attribute__((amdgpu_waves_per_eu(2)))
void GCU_nbr_kernel(
    const float* __restrict__ z,   // [N,128]
    const float* __restrict__ zn,  // [N,32,128]
    const float* __restrict__ w1,  // [128,256]
    const float* __restrict__ b1,  // [128]
    const float* __restrict__ w2,  // [128,128]
    const float* __restrict__ b2,  // [128]
    float* __restrict__ vsum,      // [N,128] out
    int N) {
  __shared__ __align__(16) _Float16 w1h[128 * 256];   // 64 KB
  __shared__ __align__(16) _Float16 w2h[128 * 128];   // 32 KB
  __shared__ __align__(16) _Float16 hst[8 * 16 * HS]; // 34 KB per-wave stage

  const int tid = threadIdx.x;
#pragma unroll 4
  for (int i = tid; i < 128 * 256; i += 256) w1h[i] = (_Float16)w1[i];
#pragma unroll 2
  for (int i = tid; i < 128 * 128; i += 256) w2h[i] = (_Float16)w2[i];
  __syncthreads();

  const int wave = tid >> 5;
  const int lane = tid & 31;
  const int node = blockIdx.x * 8 + wave;
  if (node >= N) return;

  const int nn = lane & 15;                 // C/B column within tile
  const int rbase = (lane & 16) ? 8 : 0;    // C row base
  _Float16* hrow = hst + wave * 16 * HS;    // this wave's [16][HS] stage

  float ssum[8];
#pragma unroll
  for (int nt = 0; nt < 8; ++nt) ssum[nt] = 0.f;

  const float* zrow = z + (size_t)node * DIM_D;

#pragma unroll
  for (int mt = 0; mt < 2; ++mt) {          // two 16-neighbor row tiles
    // ---- layer 1: [16 x 256] @ w1^T[256 x 128] ----
    v16h a1[8];
#pragma unroll
    for (int kk = 0; kk < 8; ++kk) {
      if (kk < 4) {                         // zc cols 0..127: z[node]
        a1[kk] = load_a_f32(zrow, kk * 32, lane);
      } else {                              // zc cols 128..255: z_[node,k]
        const float* r =
            zn + ((size_t)node * KNBR + mt * 16 + (lane & 15)) * DIM_D;
        a1[kk] = load_a_f32(r, (kk - 4) * 32, lane);
      }
    }
    // nt-outer: one live accumulator, epilogue immediately
#pragma unroll
    for (int nt = 0; nt < 8; ++nt) {
      v8f acc = vzero8();
#pragma unroll
      for (int kk = 0; kk < 8; ++kk) {
        v16h b = load_b_lds(w1h, 256, nt * 16, kk * 32, lane);
        acc = wmma_f16(a1[kk], b, acc);
      }
      const int col = nt * 16 + nn;
      const float bias = b1[col];
#pragma unroll
      for (int r = 0; r < 8; ++r)
        hrow[(rbase + r) * HS + col] = (_Float16)fast_tanh(acc[r] + bias);
    }

    // ---- layer 2: [16 x 128] @ w2^T[128 x 128], tanh, row-sum ----
    v16h a2[4];
#pragma unroll
    for (int kk = 0; kk < 4; ++kk)
      a2[kk] = load_a_lds(hrow + (lane & 15) * HS, kk * 32, lane);
#pragma unroll
    for (int nt = 0; nt < 8; ++nt) {
      v8f acc = vzero8();
#pragma unroll
      for (int kk = 0; kk < 4; ++kk) {
        v16h b = load_b_lds(w2h, 128, nt * 16, kk * 32, lane);
        acc = wmma_f16(a2[kk], b, acc);
      }
      const float bias = b2[nt * 16 + nn];
      float s = 0.f;
#pragma unroll
      for (int r = 0; r < 8; ++r) s += fast_tanh(acc[r] + bias);
      ssum[nt] += s;                        // sum over this tile's 8 rows
    }
  }

  // combine lane-half partials (rows 0-7 <-> 8-15) and store v_ row
#pragma unroll
  for (int nt = 0; nt < 8; ++nt) {
    float s = ssum[nt] + __shfl_xor(ssum[nt], 16, 32);
    if (lane < 16) vsum[(size_t)node * DIM_H + nt * 16 + lane] = s;
  }
}

// ===========================================================================
// Kernel 2: current branch + output head.  One wave per 16 nodes.
// ===========================================================================
__global__ __launch_bounds__(256)
__attribute__((amdgpu_waves_per_eu(2)))
void GCU_node_kernel(
    const float* __restrict__ z,    // [N,128]
    const float* __restrict__ cw1,  // [128,128]
    const float* __restrict__ cb1,
    const float* __restrict__ cw2,  // [128,128]
    const float* __restrict__ cb2,
    const float* __restrict__ ow,   // [64,256]
    const float* __restrict__ ob,   // [64]
    const float* __restrict__ vsum, // [N,128]
    float* __restrict__ out,        // [N,64]
    int ntiles) {
  __shared__ __align__(16) _Float16 w1h[128 * 128];
  __shared__ __align__(16) _Float16 w2h[128 * 128];
  __shared__ __align__(16) _Float16 owh[64 * 256];
  __shared__ __align__(16) _Float16 hst[8 * 16 * HS];

  const int tid = threadIdx.x;
#pragma unroll 2
  for (int i = tid; i < 128 * 128; i += 256) w1h[i] = (_Float16)cw1[i];
#pragma unroll 2
  for (int i = tid; i < 128 * 128; i += 256) w2h[i] = (_Float16)cw2[i];
#pragma unroll 2
  for (int i = tid; i < 64 * 256; i += 256) owh[i] = (_Float16)ow[i];
  __syncthreads();

  const int wave = tid >> 5;
  const int lane = tid & 31;
  const int tile = blockIdx.x * 8 + wave;
  if (tile >= ntiles) return;

  const int m0 = tile * 16;
  const int nn = lane & 15;
  const int rbase = (lane & 16) ? 8 : 0;
  _Float16* hrow = hst + wave * 16 * HS;

  // ---- layer 1: [16 x 128] @ cw1^T ----
  v16h a1[4];
#pragma unroll
  for (int kk = 0; kk < 4; ++kk) {
    const float* r = z + (size_t)(m0 + (lane & 15)) * DIM_D;
    a1[kk] = load_a_f32(r, kk * 32, lane);
  }
#pragma unroll
  for (int nt = 0; nt < 8; ++nt) {
    v8f acc = vzero8();
#pragma unroll
    for (int kk = 0; kk < 4; ++kk) {
      v16h b = load_b_lds(w1h, 128, nt * 16, kk * 32, lane);
      acc = wmma_f16(a1[kk], b, acc);
    }
    const int col = nt * 16 + nn;
    const float bias = cb1[col];
#pragma unroll
    for (int r = 0; r < 8; ++r)
      hrow[(rbase + r) * HS + col] = (_Float16)fast_tanh(acc[r] + bias);
  }

  // ---- layer 2: [16 x 128] @ cw2^T ----
  v16h a2[4];
#pragma unroll
  for (int kk = 0; kk < 4; ++kk)
    a2[kk] = load_a_lds(hrow + (lane & 15) * HS, kk * 32, lane);
  // a2 is in registers now; the per-nt epilogue may safely overwrite hrow.
#pragma unroll
  for (int nt = 0; nt < 8; ++nt) {
    v8f acc = vzero8();
#pragma unroll
    for (int kk = 0; kk < 4; ++kk) {
      v16h b = load_b_lds(w2h, 128, nt * 16, kk * 32, lane);
      acc = wmma_f16(a2[kk], b, acc);
    }
    const int col = nt * 16 + nn;
    const float bias = cb2[col];
#pragma unroll
    for (int r = 0; r < 8; ++r)
      hrow[(rbase + r) * HS + col] = (_Float16)fast_tanh(acc[r] + bias);
  }

  // ---- output head: [16 x 256] @ ow^T[256 x 64]; A = [v | v_] ----
  v16h a3[8];
#pragma unroll
  for (int kk = 0; kk < 8; ++kk) {
    if (kk < 4) {
      a3[kk] = load_a_lds(hrow + (lane & 15) * HS, kk * 32, lane);  // v (f16)
    } else {
      const float* r = vsum + (size_t)(m0 + (lane & 15)) * DIM_H;
      a3[kk] = load_a_f32(r, (kk - 4) * 32, lane);                  // v_ (f32)
    }
  }
#pragma unroll
  for (int nt = 0; nt < 4; ++nt) {
    v8f acc = vzero8();
#pragma unroll
    for (int kk = 0; kk < 8; ++kk) {
      v16h b = load_b_lds(owh, 256, nt * 16, kk * 32, lane);
      acc = wmma_f16(a3[kk], b, acc);
    }
    const int col = nt * 16 + nn;
    const float bias = ob[col];
#pragma unroll
    for (int r = 0; r < 8; ++r)
      out[(size_t)(m0 + rbase + r) * DIM_C + col] = acc[r] + bias;
  }
}

// ===========================================================================
extern "C" void kernel_launch(void* const* d_in, const int* in_sizes, int n_in,
                              void* d_out, int out_size, void* d_ws,
                              size_t ws_size, hipStream_t stream) {
  const float* z   = (const float*)d_in[0];
  const float* zn  = (const float*)d_in[1];
  const float* cw1 = (const float*)d_in[2];
  const float* cb1 = (const float*)d_in[3];
  const float* cw2 = (const float*)d_in[4];
  const float* cb2 = (const float*)d_in[5];
  const float* nw1 = (const float*)d_in[6];
  const float* nb1 = (const float*)d_in[7];
  const float* nw2 = (const float*)d_in[8];
  const float* nb2 = (const float*)d_in[9];
  const float* ow  = (const float*)d_in[10];
  const float* ob  = (const float*)d_in[11];
  float* out = (float*)d_out;

  const int N = in_sizes[0] / DIM_D;      // 20000
  float* vsum = (float*)d_ws;             // [N,128] f32 scratch

  const int blocks1 = (N + 7) / 8;        // one wave per node
  GCU_nbr_kernel<<<blocks1, 256, 0, stream>>>(z, zn, nw1, nb1, nw2, nb2,
                                              vsum, N);

  const int ntiles = (N + 15) / 16;       // one wave per 16 nodes
  const int blocks2 = (ntiles + 7) / 8;
  GCU_node_kernel<<<blocks2, 256, 0, stream>>>(z, cw1, cb1, cw2, cb2, ow, ob,
                                               vsum, out, ntiles);
}
